// CharRNN_1297080123701
// MI455X (gfx1250) — matrix operand order repaired and checked by reading, not compiled, not run on hardware
//
#include <hip/hip_runtime.h>
#include <hip/hip_bf16.h>
#include <math.h>

// ---------------------------------------------------------------------------
// CharRNN forward for gfx1250 (MI455X), wave32 + WMMA bf16 + TDM staging.
//   d_in: [seq i32 TxB][emb f32][wax f32][waa f32][waa_b f32][wya f32][wya_b f32]
//   d_out: out f32 [T,B,256] ++ h_last f32 [B,512]
// ---------------------------------------------------------------------------

#define TT   2048
#define BB   64
#define EMBD 128
#define HID  512
#define NCH  256
#define M1   (TT * BB)          // 131072 rows of the big GEMMs
#define KLDS 256                // K 0..255 of Waa lives in LDS (256KB)
#define LDS_BYTES (BB * HID * 2 + HID * KLDS * 2)   // 64KB + 256KB = 320KB

typedef __attribute__((ext_vector_type(16))) __bf16 bf16x16;
typedef __attribute__((ext_vector_type(8)))  float  f32x8;
typedef __attribute__((ext_vector_type(4)))  unsigned int u32x4;
typedef __attribute__((ext_vector_type(4)))  int    i32x4;
typedef __attribute__((ext_vector_type(8)))  int    i32x8;

#if __has_builtin(__builtin_amdgcn_tensor_load_to_lds) && \
    __has_builtin(__builtin_amdgcn_s_wait_tensorcnt)
#define HAVE_TDM 1
#endif

// ---- WMMA fragment helpers (16x16x32 bf16, wave32 layouts per CDNA5 ISA) ---
// A (16x32, MxK): lanes 0-15 row M=lane hold K {0..7,16..23}; lanes 16-31 same
// rows hold K {8..15,24..31}.  rowbase = &A[row_of_this_lane][0].
__device__ __forceinline__ bf16x16 ldA(const __bf16* rowbase, int k0, int half) {
    bf16x16 r;
    const __bf16* p = rowbase + k0 + half * 8;
    *(uint4*)&r       = *(const uint4*)(p);
    *((uint4*)&r + 1) = *(const uint4*)(p + 16);
    return r;
}
// B (32x16, KxN) from row-major W[n][k]: lane (n = lane&15) holds K
// {0..7,16..23} (lanes 0-15) or {8..15,24..31} (lanes 16-31), matching A.
__device__ __forceinline__ bf16x16 ldB(const __bf16* rowbase, int k0, int half) {
    bf16x16 r;
    const __bf16* p = rowbase + k0 + half * 8;
    *(uint4*)&r       = *(const uint4*)(p);
    *((uint4*)&r + 1) = *(const uint4*)(p + 16);
    return r;
}
__device__ __forceinline__ f32x8 wmma_bf16(bf16x16 a, bf16x16 b, f32x8 c) {
    return __builtin_amdgcn_wmma_f32_16x16x32_bf16(
        /*neg_a=*/false, a, /*neg_b=*/false, b,
        /*c_mod=*/(short)0, c, /*reuse_a=*/false, /*reuse_b=*/false);
}

// ---------------------------------------------------------------------------
// Kernel 0: convert weights/embedding f32 -> bf16 scratch copies.
// ---------------------------------------------------------------------------
__global__ __launch_bounds__(256) void k_cvt(
    const float* __restrict__ emb, const float* __restrict__ wax,
    const float* __restrict__ waa, const float* __restrict__ wya,
    __bf16* __restrict__ emb_bf, __bf16* __restrict__ wax_bf,
    __bf16* __restrict__ waa_bf, __bf16* __restrict__ wya_bf) {
    int i = blockIdx.x * 256 + threadIdx.x;
    if (i < NCH * EMBD) emb_bf[i] = (__bf16)emb[i];
    if (i < HID * EMBD) wax_bf[i] = (__bf16)wax[i];
    if (i < HID * HID)  waa_bf[i] = (__bf16)waa[i];
    if (i < NCH * HID)  wya_bf[i] = (__bf16)wya[i];
}

// ---------------------------------------------------------------------------
// Kernel 1: xw[m][h] = sum_e emb_bf[seq[m]][e] * wax_bf[h][e]   (M1 x 512)
// One 16x16 output tile per wave; K = 128 -> 4 WMMAs.
// grid = (M1/16, 4), block = 256 (8 waves -> 8 N-tiles per block).
// ---------------------------------------------------------------------------
__global__ __launch_bounds__(256) void k_embed_proj(
    const int* __restrict__ seq, const __bf16* __restrict__ emb_bf,
    const __bf16* __restrict__ wax_bf, float* __restrict__ xw) {
    const int lane = threadIdx.x & 31;
    const int wave = threadIdx.x >> 5;
    const int half = lane >> 4, nl = lane & 15;
    const int m0 = blockIdx.x * 16;
    const int n0 = (blockIdx.y * 8 + wave) * 16;

    const int c = seq[m0 + nl];                      // embedding row for this lane's A row
    const __bf16* arow = emb_bf + (size_t)c * EMBD;
    const __bf16* brow = wax_bf + (size_t)(n0 + nl) * EMBD;

    f32x8 acc = {};
#pragma unroll
    for (int kt = 0; kt < EMBD / 32; ++kt)
        acc = wmma_bf16(ldA(arow, kt * 32, half), ldB(brow, kt * 32, half), acc);

    float* orow = xw + (size_t)m0 * HID + n0 + nl;
#pragma unroll
    for (int r = 0; r < 8; ++r)
        orow[(size_t)(r + half * 8) * HID] = acc[r];  // C layout: VGPR r -> row r(+8)
}

// ---------------------------------------------------------------------------
// Kernel 2: sequential recurrence on ONE WGP (critical path).
//   1024 threads = 32 waves; wave w owns output columns [w*16, w*16+16).
//   LDS (320KB = WGP max): h single buffer 64KB | Waa[K 0..255] 256KB.
//     Waa LDS half staged once via the Tensor Data Mover (tensor_load_to_lds
//     + s_wait_tensorcnt) when available.
//   Waa[K 256..511] register-resident: 8 frags x 8 VGPRs / lane.
//   Two barriers/step: reads(h_t) -> barrier -> in-place write(h_{t+1}) -> barrier.
// ---------------------------------------------------------------------------
__global__ __launch_bounds__(1024) void k_recurrence(
    const float* __restrict__ xw, const float* __restrict__ waa_b,
    const __bf16* __restrict__ waa_bf, __bf16* __restrict__ hs,
    float* __restrict__ h_last) {
    extern __shared__ char smem[];
    __bf16* hbuf = (__bf16*)smem;                      // [64][512]  bf16, 64KB
    __bf16* wls  = (__bf16*)(smem + BB * HID * 2);     // [512][256] bf16, 256KB

    const int tid  = threadIdx.x;
    const int lane = tid & 31, wave = tid >> 5;
    const int half = lane >> 4, nl = lane & 15;
    const int n0 = wave * 16;
    const int n  = n0 + nl;

    // ---- stage Waa[ : , 0:256 ] into LDS ----
#if defined(HAVE_TDM)
    if (wave == 0) {
        // Tensor DMA descriptor (D#): 2D tensor 512x512 bf16, tile 512x256,
        // row stride 512 elements -> LDS rows of 256 contiguous elements.
        unsigned long long ga = (unsigned long long)(uintptr_t)waa_bf;
        unsigned lds_base = (unsigned)(uintptr_t)wls;       // flat LDS addr[31:0]
        u32x4 g0;
        g0[0] = 1u;                                         // count=1, user D#
        g0[1] = lds_base;                                   // lds_addr
        g0[2] = (unsigned)ga;                               // global_addr[31:0]
        g0[3] = (unsigned)(ga >> 32) | (2u << 30);          // addr[56:32] | type=2
        i32x8 g1;
        g1[0] = (int)(1u << 16);                            // data_size = 2 bytes
        g1[1] = (int)(512u << 16);                          // tensor_dim0 = 512
        g1[2] = (int)(512u << 16);                          // tensor_dim1 = 512
        g1[3] = (int)(256u << 16);                          // tile_dim0   = 256
        g1[4] = 512;                                        // tile_dim1   = 512
        g1[5] = 512;                                        // dim0_stride = 512
        g1[6] = 0; g1[7] = 0;
        i32x4 z4 = {0, 0, 0, 0};
#if __clang_major__ >= 23
        i32x8 z8 = {0, 0, 0, 0, 0, 0, 0, 0};
        __builtin_amdgcn_tensor_load_to_lds(g0, g1, z4, z4, z8, 0);
#else
        __builtin_amdgcn_tensor_load_to_lds(g0, g1, z4, z4, 0);
#endif
        __builtin_amdgcn_s_wait_tensorcnt(0);
    }
#else
    for (int i = tid; i < (HID * KLDS) / 8; i += 1024) {
        const int e = i * 8;                   // 8 bf16 = 16B per iteration
        const int r = e >> 8, c = e & (KLDS - 1);
        *(uint4*)(wls + e) = *(const uint4*)(waa_bf + (size_t)r * HID + c);
    }
#endif

    // ---- register-resident Waa[ n , 256:512 ] fragments for this wave ----
    bf16x16 bfrag[(HID - KLDS) / 32];
    const __bf16* wrow = waa_bf + (size_t)n * HID;
#pragma unroll
    for (int kt = 0; kt < (HID - KLDS) / 32; ++kt)
        bfrag[kt] = ldB(wrow, KLDS + kt * 32, half);
    const float bias = waa_b[n];

    // h0 = 0  (16B stores)
    const uint4 zero4 = make_uint4(0u, 0u, 0u, 0u);
    for (int i = tid; i < (BB * HID) / 8; i += 1024)
        *(uint4*)(hbuf + i * 8) = zero4;
    __syncthreads();

    for (int t = 0; t < TT; ++t) {
        const float* xwt = xw + (size_t)t * (BB * HID);

        // acc init = xw_t + bias  (C layout: lane col n, rows r + 8*half per m-tile)
        f32x8 acc[4];
#pragma unroll
        for (int mt = 0; mt < 4; ++mt)
#pragma unroll
            for (int r = 0; r < 8; ++r)
                acc[mt][r] = xwt[(size_t)(mt * 16 + r + half * 8) * HID + n] + bias;

        // Warm next step's xw tile (128KB) into the cache hierarchy.
        if (t + 1 < TT)
            __builtin_prefetch(xwt + BB * HID + tid * 32, 0, 1);  // 1024 lanes x 128B

        // K 0..255: B fragments from LDS-resident Waa
#pragma unroll
        for (int kt = 0; kt < KLDS / 32; ++kt) {
            const bf16x16 bl = ldB(wls + (size_t)n * KLDS, kt * 32, half);
#pragma unroll
            for (int mt = 0; mt < 4; ++mt) {
                bf16x16 a = ldA(hbuf + (size_t)(mt * 16 + nl) * HID, kt * 32, half);
                acc[mt] = wmma_bf16(a, bl, acc[mt]);
            }
        }
        // K 256..511: B fragments from VGPRs
#pragma unroll
        for (int kt = 0; kt < (HID - KLDS) / 32; ++kt)
#pragma unroll
            for (int mt = 0; mt < 4; ++mt) {
                bf16x16 a = ldA(hbuf + (size_t)(mt * 16 + nl) * HID, KLDS + kt * 32, half);
                acc[mt] = wmma_bf16(a, bfrag[kt], acc[mt]);
            }

        __syncthreads();   // all reads of h_t complete -> safe to overwrite in place

        // tanh, write h_{t+1} in place + hs (bf16) to global.
        __bf16* hsrow = hs + (size_t)t * (BB * HID);
#pragma unroll
        for (int mt = 0; mt < 4; ++mt)
#pragma unroll
            for (int r = 0; r < 8; ++r) {
                const int m = mt * 16 + r + half * 8;
                const float hv = tanhf(acc[mt][r]);
                const __bf16 hb = (__bf16)hv;
                hbuf[m * HID + n] = hb;
                hsrow[(size_t)m * HID + n] = hb;
                if (t == TT - 1) h_last[(size_t)m * HID + n] = hv;
            }
        __syncthreads();   // h_{t+1} visible to all waves
    }
}

// ---------------------------------------------------------------------------
// Kernel 3: out[m][v] = sum_h hs[m][h] * wya_bf[v][h] + wya_b[v]
// grid = (M1/16, 2), block = 256 (8 waves -> 16 N-tiles of 256 outputs).
// ---------------------------------------------------------------------------
__global__ __launch_bounds__(256) void k_out_proj(
    const __bf16* __restrict__ hs, const __bf16* __restrict__ wya_bf,
    const float* __restrict__ wya_b, float* __restrict__ out) {
    const int lane = threadIdx.x & 31;
    const int wave = threadIdx.x >> 5;
    const int half = lane >> 4, nl = lane & 15;
    const int m0 = blockIdx.x * 16;
    const int n0 = (blockIdx.y * 8 + wave) * 16;

    const __bf16* arow = hs + (size_t)(m0 + nl) * HID;
    const __bf16* brow = wya_bf + (size_t)(n0 + nl) * HID;

    const float bias = wya_b[n0 + nl];   // C layout: lane's column is n0+nl for all rows
    f32x8 acc;
#pragma unroll
    for (int r = 0; r < 8; ++r) acc[r] = bias;

#pragma unroll
    for (int kt = 0; kt < HID / 32; ++kt)
        acc = wmma_bf16(ldA(arow, kt * 32, half), ldB(brow, kt * 32, half), acc);

    float* orow = out + (size_t)m0 * NCH + n0 + nl;
#pragma unroll
    for (int r = 0; r < 8; ++r)
        orow[(size_t)(r + half * 8) * NCH] = acc[r];
}

// ---------------------------------------------------------------------------
extern "C" void kernel_launch(void* const* d_in, const int* in_sizes, int n_in,
                              void* d_out, int out_size, void* d_ws, size_t ws_size,
                              hipStream_t stream) {
    const int*   seq   = (const int*)d_in[0];
    const float* emb   = (const float*)d_in[1];
    const float* wax   = (const float*)d_in[2];
    const float* waa   = (const float*)d_in[3];
    const float* waa_b = (const float*)d_in[4];
    const float* wya   = (const float*)d_in[5];
    const float* wya_b = (const float*)d_in[6];
    float* out = (float*)d_out;

    // Workspace layout (bytes):
    //   xw   f32  [M1][512]   256 MB
    //   hs   bf16 [M1][512]   128 MB
    //   emb/wax/waa/wya bf16 copies (~1 MB)
    char* ws = (char*)d_ws;
    size_t off = 0;
    float*  xw     = (float*)(ws + off);  off += (size_t)M1 * HID * sizeof(float);
    __bf16* hs     = (__bf16*)(ws + off); off += (size_t)M1 * HID * sizeof(__bf16);
    __bf16* emb_bf = (__bf16*)(ws + off); off += (size_t)NCH * EMBD * sizeof(__bf16);
    __bf16* wax_bf = (__bf16*)(ws + off); off += (size_t)HID * EMBD * sizeof(__bf16);
    __bf16* waa_bf = (__bf16*)(ws + off); off += (size_t)HID * HID * sizeof(__bf16);
    __bf16* wya_bf = (__bf16*)(ws + off); off += (size_t)NCH * HID * sizeof(__bf16);
    (void)off; (void)in_sizes; (void)n_in; (void)out_size; (void)ws_size;

    // Allow 320KB dynamic LDS for the recurrence kernel (immediate call; not a
    // stream op, so graph capture is unaffected).
    (void)hipFuncSetAttribute(reinterpret_cast<const void*>(k_recurrence),
                              hipFuncAttributeMaxDynamicSharedMemorySize,
                              LDS_BYTES);

    // 0) f32 -> bf16 weight copies (largest tensor: waa = 262144 elements)
    k_cvt<<<(HID * HID + 255) / 256, 256, 0, stream>>>(
        emb, wax, waa, wya, emb_bf, wax_bf, waa_bf, wya_bf);

    // 1) xw = emb[seq] @ Wax^T   (whole-chip WMMA)
    k_embed_proj<<<dim3(M1 / 16, (HID / 16) / 8), 256, 0, stream>>>(
        seq, emb_bf, wax_bf, xw);

    // 2) sequential recurrence: 1 workgroup, 32 waves, 320KB LDS
    k_recurrence<<<1, 1024, LDS_BYTES, stream>>>(
        xw, waa_b, waa_bf, hs, out + (size_t)M1 * NCH);

    // 3) out = hs @ Wya^T + b    (whole-chip WMMA)
    k_out_proj<<<dim3(M1 / 16, (NCH / 16) / 8), 256, 0, stream>>>(
        hs, wya_bf, wya_b, out);
}
